// MemexQA_19774029431002
// MI455X (gfx1250) — matrix operand order repaired and checked by reading, not compiled, not run on hardware
//
#include <hip/hip_runtime.h>

// ---------------------------------------------------------------------------
// MemexQA forward on gfx1250 (CDNA5): bf16 WMMA GEMMs + fused flash attention.
// Rev 3: ping-pong double-buffered GEMM (no register rotation copies),
//        pre-resolved lane pointers, K-fragment pipelining in attention.
// ---------------------------------------------------------------------------

typedef __bf16 bf16_t;
typedef __attribute__((ext_vector_type(16))) __bf16 v16bf;
typedef __attribute__((ext_vector_type(8)))  __bf16 v8bf;
typedef __attribute__((ext_vector_type(8)))  float  v8f;

#define NEGV (-1e30f)

static __device__ __forceinline__ v8f v8f_zero() {
  v8f z = {0.f, 0.f, 0.f, 0.f, 0.f, 0.f, 0.f, 0.f};
  return z;
}

// A fragment from a fully lane-resolved pointer:
// p = &A[row0 + (lane&15)][k0 + ((lane>>4)<<3)]; reads 8 + 8 contiguous bf16.
static __device__ __forceinline__ v16bf frag_A(const bf16_t* __restrict__ p) {
  v8bf lo = *(const v8bf*)p;
  v8bf hi = *(const v8bf*)(p + 16);
  return __builtin_shufflevector(lo, hi, 0,1,2,3,4,5,6,7,8,9,10,11,12,13,14,15);
}

// B fragment from a fully lane-resolved pointer into a transposed [N,K] store:
// p = &Bt[n0 + (lane&15)][k0 + ((lane>>4)<<3)]; reads 8 + 8 contiguous bf16.
static __device__ __forceinline__ v16bf frag_B(const bf16_t* __restrict__ p) {
  v8bf lo = *(const v8bf*)p;
  v8bf hi = *(const v8bf*)(p + 16);
  return __builtin_shufflevector(lo, hi, 0,1,2,3,4,5,6,7,8,9,10,11,12,13,14,15);
}

static __device__ __forceinline__ float half_max16(float v) {
  #pragma unroll
  for (int s = 1; s < 16; s <<= 1) v = fmaxf(v, __shfl_xor(v, s, 32));
  return v;
}
static __device__ __forceinline__ float half_sum16(float v) {
  #pragma unroll
  for (int s = 1; s < 16; s <<= 1) v += __shfl_xor(v, s, 32);
  return v;
}

// --------------------------- elementwise helpers ---------------------------

__global__ void k_cvt_bf16(const float* __restrict__ in, bf16_t* __restrict__ out, int n) {
  int i = blockIdx.x * 256 + threadIdx.x;
  if (i < n) out[i] = (bf16_t)in[i];
}

// out[(h*Np + kk)*Kc + d] = in[(h*Kc + d)*Np + kk]   (weight transpose + cvt)
__global__ void k_transpose_cvt(const float* __restrict__ in, bf16_t* __restrict__ out,
                                int Kc, int Np, int total) {
  int idx = blockIdx.x * 256 + threadIdx.x;
  if (idx >= total) return;
  int d = idx % Kc;
  int rest = idx / Kc;
  int kk = rest % Np;
  int h  = rest / Np;
  out[idx] = (bf16_t)in[((size_t)(h * Kc + d)) * Np + kk];
}

// LayerNorm over last dim (1024) -> bf16
__global__ __launch_bounds__(256) void k_ln_bf16(const float* __restrict__ x,
                                                 const float* __restrict__ g,
                                                 const float* __restrict__ bta,
                                                 bf16_t* __restrict__ out, int Dd) {
  __shared__ float red[256];
  int row = blockIdx.x, tid = threadIdx.x;
  const float* xr = x + (size_t)row * Dd;
  float s = 0.f;
  for (int i = tid; i < Dd; i += 256) s += xr[i];
  red[tid] = s; __syncthreads();
  for (int st = 128; st > 0; st >>= 1) { if (tid < st) red[tid] += red[tid + st]; __syncthreads(); }
  float mean = red[0] / Dd;
  __syncthreads();
  float v = 0.f;
  for (int i = tid; i < Dd; i += 256) { float d = xr[i] - mean; v += d * d; }
  red[tid] = v; __syncthreads();
  for (int st = 128; st > 0; st >>= 1) { if (tid < st) red[tid] += red[tid + st]; __syncthreads(); }
  float rstd = rsqrtf(red[0] / Dd + 1e-5f);
  for (int i = tid; i < Dd; i += 256)
    out[(size_t)row * Dd + i] = (bf16_t)((xr[i] - mean) * rstd * g[i] + bta[i]);
}

// ------------------------------ WMMA GEMM ----------------------------------
// C[M,N] = act(A[M,K](bf16,row-major) x Bt[N,K](bf16) + bias[N]); out f32 or bf16.
// Per wave: 32x64 tile. k-loop unrolled 2x with ping-pong fragment buffers so
// loads for chunk k+1 are in flight while chunk k's WMMAs execute, with no
// register rotation copies. nchunk (K/32) is even for every K used here.
__global__ __launch_bounds__(128) void k_gemm_wmma(
    const bf16_t* __restrict__ A, const bf16_t* __restrict__ Bt,
    const float* __restrict__ bias, float* __restrict__ Cf, bf16_t* __restrict__ Cb,
    int M, int N, int K, int relu) {
  int lane = threadIdx.x & 31;
  int wave = threadIdx.x >> 5;
  int rowbase = (blockIdx.y * 2 + (wave >> 1)) * 32;
  int colbase = blockIdx.x * 128 + (wave & 1) * 64;
  if (rowbase >= M) return;

  v8f acc[2][4];
  #pragma unroll
  for (int rr = 0; rr < 2; ++rr)
    #pragma unroll
    for (int t = 0; t < 4; ++t) acc[rr][t] = v8f_zero();

  // lane-resolved base pointers (per-chunk addressing = base + ko)
  const bf16_t* pA0 = A + ((size_t)rowbase + (lane & 15)) * K + ((lane >> 4) << 3);
  const bf16_t* pA1 = pA0 + (size_t)16 * K;
  const bf16_t* pB[4];
  #pragma unroll
  for (int t = 0; t < 4; ++t)
    pB[t] = Bt + ((size_t)colbase + t * 16 + (lane & 15)) * K + ((lane >> 4) << 3);

  int nchunk = K >> 5;  // even (32 or 64)

  // buffer X <- chunk 0
  v16bf aX0 = frag_A(pA0), aX1 = frag_A(pA1);
  v16bf bX[4];
  #pragma unroll
  for (int t = 0; t < 4; ++t) bX[t] = frag_B(pB[t]);

  v16bf aY0, aY1, bY[4];
  for (int kc = 0; kc < nchunk; kc += 2) {
    // buffer Y <- chunk kc+1 (always exists: nchunk even)
    {
      int ko = (kc + 1) << 5;
      if (kc + 2 < nchunk) __builtin_prefetch(pA0 + ((kc + 2) << 5), 0, 3);
      aY0 = frag_A(pA0 + ko);
      aY1 = frag_A(pA1 + ko);
      #pragma unroll
      for (int t = 0; t < 4; ++t) bY[t] = frag_B(pB[t] + ko);
    }
    // WMMAs on buffer X (chunk kc)
    #pragma unroll
    for (int t = 0; t < 4; ++t) {
      acc[0][t] = __builtin_amdgcn_wmma_f32_16x16x32_bf16(false, aX0, false, bX[t],
                                                          (short)0, acc[0][t], false, false);
      acc[1][t] = __builtin_amdgcn_wmma_f32_16x16x32_bf16(false, aX1, false, bX[t],
                                                          (short)0, acc[1][t], false, false);
    }
    // buffer X <- chunk kc+2 (clamped on final trip; redundant reload is harmless)
    {
      int kn = (kc + 2 < nchunk) ? (kc + 2) : kc;
      int ko = kn << 5;
      aX0 = frag_A(pA0 + ko);
      aX1 = frag_A(pA1 + ko);
      #pragma unroll
      for (int t = 0; t < 4; ++t) bX[t] = frag_B(pB[t] + ko);
    }
    // WMMAs on buffer Y (chunk kc+1)
    #pragma unroll
    for (int t = 0; t < 4; ++t) {
      acc[0][t] = __builtin_amdgcn_wmma_f32_16x16x32_bf16(false, aY0, false, bY[t],
                                                          (short)0, acc[0][t], false, false);
      acc[1][t] = __builtin_amdgcn_wmma_f32_16x16x32_bf16(false, aY1, false, bY[t],
                                                          (short)0, acc[1][t], false, false);
    }
  }

  int hi = lane >> 4, col0 = lane & 15;
  #pragma unroll
  for (int rr = 0; rr < 2; ++rr) {
    #pragma unroll
    for (int t = 0; t < 4; ++t) {
      int col = colbase + t * 16 + col0;
      float bv = bias ? bias[col] : 0.f;
      #pragma unroll
      for (int r = 0; r < 8; ++r) {
        int row = rowbase + rr * 16 + r + 8 * hi;
        float v = acc[rr][t][r] + bv;
        if (relu) v = fmaxf(v, 0.f);
        if (Cf) Cf[(size_t)row * N + col] = v;
        else    Cb[(size_t)row * N + col] = (bf16_t)v;
      }
    }
  }
}

// -------------------------- scatter / mask kernels -------------------------
// temp[row, h*128+k] (f32, row = b*T + l) -> dst[b,h, loff+l, k] (bf16, zeroed pad)
__global__ void k_scatter_qk(const float* __restrict__ temp, bf16_t* __restrict__ dst,
                             const int* __restrict__ len, int T, int loff, int total) {
  int idx = blockIdx.x * 256 + threadIdx.x;
  if (idx >= total) return;
  int c = idx & 1023, row = idx >> 10;
  int b = row / T, ll = row % T;
  int h = c >> 7, k = c & 127;
  float v = (ll < len[b]) ? temp[idx] : 0.f;
  dst[(((size_t)b * 8 + h) * 576 + loff + ll) * 128 + k] = (bf16_t)v;
}

// V transposed: dst[b,h, v, loff+l] (bf16, zeroed pad); dst leading dim = 576
__global__ void k_scatter_vT(const float* __restrict__ temp, bf16_t* __restrict__ dst,
                             const int* __restrict__ len, int T, int loff, int total) {
  int idx = blockIdx.x * 256 + threadIdx.x;
  if (idx >= total) return;
  int c = idx & 1023, row = idx >> 10;
  int b = row / T, ll = row % T;
  int h = c >> 7, k = c & 127;
  float v = (ll < len[b]) ? temp[idx] : 0.f;
  dst[(((size_t)b * 8 + h) * 128 + k) * 576 + loff + ll] = (bf16_t)v;
}

// zero rows of x[B, 576, 1024] at invalid sequence positions
__global__ void k_mask_seq(float* __restrict__ x, const int* __restrict__ tlen,
                           const int* __restrict__ ilen, int total) {
  int idx = blockIdx.x * 256 + threadIdx.x;
  if (idx >= total) return;
  int l = (idx >> 10) % 576;
  int b = idx / (576 * 1024);
  bool valid = (l < 512) ? (l < tlen[b]) : ((l - 512) < ilen[b]);
  if (!valid) x[idx] = 0.f;
}

// ------------------------- fused flash attention ---------------------------
// grid.x = B*H (256), grid.y = 36 row-blocks; one wave per block.
// K fragments are single-buffer pipelined: consumed by QK^T WMMAs, then
// refilled for the next j-step while softmax / LDS / PV WMMAs execute.
__global__ __launch_bounds__(32) void k_attn_fused(
    const bf16_t* __restrict__ Q, const bf16_t* __restrict__ Km,
    const bf16_t* __restrict__ Vt, bf16_t* __restrict__ feat) {
  __shared__ float pl[16 * 32];
  int lane = threadIdx.x;
  int bh = blockIdx.x;
  int b = bh >> 3, h = bh & 7;
  int i0 = blockIdx.y * 16;
  int hi = lane >> 4, col = lane & 15;

  // lane-resolved base pointers
  const bf16_t* pQ = Q + ((size_t)(bh * 576 + i0) + (lane & 15)) * 128 + ((lane >> 4) << 3);
  const bf16_t* pK = Km + ((size_t)bh * 576 + (lane & 15)) * 128 + ((lane >> 4) << 3);
  const bf16_t* pV = Vt + ((size_t)bh * 128 + (lane & 15)) * 576 + ((lane >> 4) << 3);

  v16bf aQ[4];
  #pragma unroll
  for (int c = 0; c < 4; ++c) aQ[c] = frag_A(pQ + c * 32);

  v8f O[8];
  #pragma unroll
  for (int t = 0; t < 8; ++t) O[t] = v8f_zero();
  float mrow[8], lrow[8];
  #pragma unroll
  for (int r = 0; r < 8; ++r) { mrow[r] = NEGV; lrow[r] = 0.f; }

  // prologue: K fragments for j0 = 0   (bk[c] -> cols j0..j0+15, bk[4+c] -> +16)
  v16bf bk[8];
  #pragma unroll
  for (int c = 0; c < 4; ++c) {
    bk[c]     = frag_B(pK + c * 32);
    bk[4 + c] = frag_B(pK + 16 * 128 + c * 32);
  }

  for (int j0 = 0; j0 < 576; j0 += 32) {
    // V fragments for this step: issue early, consumed at the end of the step
    v16bf bv[8];
    #pragma unroll
    for (int t2 = 0; t2 < 8; ++t2)
      bv[t2] = frag_B(pV + (size_t)(t2 * 16) * 576 + j0);

    v8f s0 = v8f_zero(), s1 = v8f_zero();
    #pragma unroll
    for (int c = 0; c < 4; ++c) {
      s0 = __builtin_amdgcn_wmma_f32_16x16x32_bf16(false, aQ[c], false, bk[c],
                                                   (short)0, s0, false, false);
      s1 = __builtin_amdgcn_wmma_f32_16x16x32_bf16(false, aQ[c], false, bk[4 + c],
                                                   (short)0, s1, false, false);
    }

    // refill K fragments for the next step (hidden under softmax + LDS + PV)
    {
      int jn = (j0 + 32 < 576) ? (j0 + 32) : j0;
      const bf16_t* pKn = pK + (size_t)jn * 128;
      #pragma unroll
      for (int c = 0; c < 4; ++c) {
        bk[c]     = frag_B(pKn + c * 32);
        bk[4 + c] = frag_B(pKn + 16 * 128 + c * 32);
      }
    }

    // mask (==0 or diagonal) + online softmax update
    float p0[8], p1[8];
    #pragma unroll
    for (int r = 0; r < 8; ++r) {
      int ig = i0 + r + 8 * hi;
      float v0 = s0[r]; if (v0 == 0.f || ig == (j0 + col))      v0 = NEGV;
      float v1 = s1[r]; if (v1 == 0.f || ig == (j0 + 16 + col)) v1 = NEGV;
      float t = half_max16(fmaxf(v0, v1));
      float mn = fmaxf(mrow[r], t);
      float sc = __expf(mrow[r] - mn);
      float e0 = __expf(v0 - mn), e1 = __expf(v1 - mn);
      float srow = half_sum16(e0 + e1);
      lrow[r] = lrow[r] * sc + srow;
      mrow[r] = mn;
      p0[r] = e0; p1[r] = e1;
      #pragma unroll
      for (int t2 = 0; t2 < 8; ++t2) O[t2][r] *= sc;
    }
    // stage P (16x32 f32) in LDS, re-gather in WMMA A layout as bf16
    #pragma unroll
    for (int r = 0; r < 8; ++r) {
      pl[(r + 8 * hi) * 32 + col]      = p0[r];
      pl[(r + 8 * hi) * 32 + 16 + col] = p1[r];
    }
    asm volatile("s_wait_dscnt 0" ::: "memory");
    v16bf aP;
    {
      const float* pp = pl + (lane & 15) * 32 + ((lane >> 4) << 3);
      #pragma unroll
      for (int e = 0; e < 8; ++e) { aP[e] = (bf16_t)pp[e]; aP[8 + e] = (bf16_t)pp[16 + e]; }
    }
    // O += P @ V   (V fragments already in registers)
    #pragma unroll
    for (int t2 = 0; t2 < 8; ++t2)
      O[t2] = __builtin_amdgcn_wmma_f32_16x16x32_bf16(false, aP, false, bv[t2],
                                                      (short)0, O[t2], false, false);
  }
  // finalize: all-masked rows -> 0 (m stuck at NEGV)
  #pragma unroll
  for (int r = 0; r < 8; ++r) {
    float inv = (mrow[r] > -1e29f && lrow[r] > 0.f) ? (1.f / lrow[r]) : 0.f;
    int ig = i0 + r + 8 * hi;
    size_t base = ((size_t)b * 576 + ig) * 1024 + h * 128;
    #pragma unroll
    for (int t2 = 0; t2 < 8; ++t2)
      feat[base + t2 * 16 + col] = (bf16_t)(O[t2][r] * inv);
  }
}

// ------------------------------ tail kernels -------------------------------

__global__ __launch_bounds__(256) void k_w2_dot(const float* __restrict__ q,
                                                const float* __restrict__ ak,
                                                float* __restrict__ w2) {
  int wv = (blockIdx.x * 256 + threadIdx.x) >> 5;
  int lane = threadIdx.x & 31;
  if (wv >= 32 * 576) return;
  int b = wv / 576;
  const float* qa = q + (size_t)b * 1024;
  const float* ar = ak + (size_t)wv * 1024;
  float s = 0.f;
  for (int i = lane; i < 1024; i += 32) s += qa[i] * ar[i];
  #pragma unroll
  for (int st = 16; st > 0; st >>= 1) s += __shfl_xor(s, st, 32);
  if (lane == 0) w2[wv] = s;
}

__global__ __launch_bounds__(256) void k_softmax_w2(float* __restrict__ w2) {
  __shared__ float red[256];
  int b = blockIdx.x, tid = threadIdx.x;
  float* row = w2 + (size_t)b * 576;
  float loc = -3.4e38f;
  for (int i = tid; i < 576; i += 256) {
    float x = row[i]; float xm = (x == 0.f) ? NEGV : x;
    loc = fmaxf(loc, xm);
  }
  red[tid] = loc; __syncthreads();
  for (int st = 128; st > 0; st >>= 1) { if (tid < st) red[tid] = fmaxf(red[tid], red[tid + st]); __syncthreads(); }
  float gmax = red[0]; __syncthreads();
  float s = 0.f;
  for (int i = tid; i < 576; i += 256) {
    float x = row[i]; float xm = (x == 0.f) ? NEGV : x;
    s += __expf(xm - gmax);
  }
  red[tid] = s; __syncthreads();
  for (int st = 128; st > 0; st >>= 1) { if (tid < st) red[tid] += red[tid + st]; __syncthreads(); }
  float gsum = red[0];
  bool any = gmax > -1e29f;
  for (int i = tid; i < 576; i += 256) {
    float x = row[i]; float xm = (x == 0.f) ? NEGV : x;
    row[i] = any ? (__expf(xm - gmax) / gsum) : 0.f;
  }
}

__global__ __launch_bounds__(256) void k_atts(const float* __restrict__ w2,
                                              const float* __restrict__ av,
                                              float* __restrict__ atts) {
  int b = blockIdx.x, tid = threadIdx.x;
  for (int v = tid; v < 1024; v += 256) {
    float s = 0.f;
    for (int l = 0; l < 576; ++l)
      s += w2[(size_t)b * 576 + l] * av[((size_t)b * 576 + l) * 1024 + v];
    atts[(size_t)b * 1024 + v] = s;
  }
}

__global__ __launch_bounds__(32) void k_final(const float* __restrict__ q,
                                              const float* __restrict__ a,
                                              const float* __restrict__ atts,
                                              const float* __restrict__ Wap,
                                              const float* __restrict__ bap,
                                              const int* __restrict__ label,
                                              float* __restrict__ out, int out_size) {
  int b = threadIdx.x;  // 32 threads
  const float* qb = q + (size_t)b * 1024;
  const float* ab = a + (size_t)b * 1024;
  const float* at = atts + (size_t)b * 1024;
  float z0 = bap[0], z1 = bap[1];
  for (int d = 0; d < 4096; ++d) {
    float c;
    if (d < 1024)      c = qb[d];
    else if (d < 2048) c = ab[d - 1024];
    else if (d < 3072) c = at[d - 2048];
    else               c = ab[d - 3072] * at[d - 3072];
    z0 += c * Wap[d * 2 + 0];
    z1 += c * Wap[d * 2 + 1];
  }
  float m = fmaxf(z0, z1);
  float e0 = __expf(z0 - m), e1 = __expf(z1 - m);
  float lse = m + __logf(e0 + e1);
  float p0 = e0 / (e0 + e1);
  int lb = label[b];
  float t0 = 0.1f + (lb == 0 ? 0.8f : 0.f);
  float t1 = 0.1f + (lb == 1 ? 0.8f : 0.f);
  float lossb = -(t0 * (z0 - lse) + t1 * (z1 - lse));
  out[b] = p0;
  #pragma unroll
  for (int st = 16; st > 0; st >>= 1) lossb += __shfl_xor(lossb, st, 32);
  if (b == 0 && out_size > 32) out[32] = lossb / 32.f;
}

// ------------------------------- host side ---------------------------------

extern "C" void kernel_launch(void* const* d_in, const int* in_sizes, int n_in,
                              void* d_out, int out_size, void* d_ws, size_t ws_size,
                              hipStream_t stream) {
  const float* question = (const float*)d_in[0];
  const float* answer   = (const float*)d_in[1];
  const float* text     = (const float*)d_in[2];
  const float* images   = (const float*)d_in[3];
  const int*   text_len = (const int*)d_in[4];
  const int*   img_len  = (const int*)d_in[5];
  const int*   label    = (const int*)d_in[6];
  const float* W_que = (const float*)d_in[7],  *b_que = (const float*)d_in[8];
  const float* W_ans = (const float*)d_in[9],  *b_ans = (const float*)d_in[10];
  const float* W_im1 = (const float*)d_in[11], *b_im1 = (const float*)d_in[12];
  const float* W_im2 = (const float*)d_in[13], *b_im2 = (const float*)d_in[14];
  const float* ln_g  = (const float*)d_in[15], *ln_b  = (const float*)d_in[16];
  const float* W_sq  = (const float*)d_in[17], *b_sq  = (const float*)d_in[18];
  const float* W_sk  = (const float*)d_in[19], *b_sk  = (const float*)d_in[20];
  const float* W_sv  = (const float*)d_in[21], *b_sv  = (const float*)d_in[22];
  const float* W_iq  = (const float*)d_in[23], *b_iq  = (const float*)d_in[24];
  const float* W_ik  = (const float*)d_in[25], *b_ik  = (const float*)d_in[26];
  const float* W_iv  = (const float*)d_in[27], *b_iv  = (const float*)d_in[28];
  const float* W_kp  = (const float*)d_in[29], *b_kp  = (const float*)d_in[30];
  const float* W_vp  = (const float*)d_in[31], *b_vp  = (const float*)d_in[32];
  const float* W_ap  = (const float*)d_in[33], *b_ap  = (const float*)d_in[34];
  float* out = (float*)d_out;

  // workspace carve-up (256B aligned)
  size_t off = 0;
  auto alloc = [&](size_t bytes) { size_t o = off; off += (bytes + 255) & ~(size_t)255; return o; };
  char* ws = (char*)d_ws;
  bf16_t* Wq_t  = (bf16_t*)(ws + alloc((size_t)1024 * 1024 * 2));
  bf16_t* Wa_t  = (bf16_t*)(ws + alloc((size_t)1024 * 1024 * 2));
  bf16_t* Wi1_t = (bf16_t*)(ws + alloc((size_t)2048 * 2048 * 2));
  bf16_t* Wi2_t = (bf16_t*)(ws + alloc((size_t)1024 * 2048 * 2));
  bf16_t* Wsq_t = (bf16_t*)(ws + alloc((size_t)1024 * 1024 * 2));
  bf16_t* Wsk_t = (bf16_t*)(ws + alloc((size_t)1024 * 1024 * 2));
  bf16_t* Wsv_t = (bf16_t*)(ws + alloc((size_t)1024 * 1024 * 2));
  bf16_t* Wiq_t = (bf16_t*)(ws + alloc((size_t)1024 * 1024 * 2));
  bf16_t* Wik_t = (bf16_t*)(ws + alloc((size_t)1024 * 1024 * 2));
  bf16_t* Wiv_t = (bf16_t*)(ws + alloc((size_t)1024 * 1024 * 2));
  bf16_t* Wkp_t = (bf16_t*)(ws + alloc((size_t)1024 * 1024 * 2));
  bf16_t* Wvp_t = (bf16_t*)(ws + alloc((size_t)1024 * 1024 * 2));
  bf16_t* qb16   = (bf16_t*)(ws + alloc((size_t)32 * 1024 * 2));
  bf16_t* ab16   = (bf16_t*)(ws + alloc((size_t)32 * 1024 * 2));
  bf16_t* imgs_b = (bf16_t*)(ws + alloc((size_t)2048 * 2048 * 2));
  bf16_t* h1_b   = (bf16_t*)(ws + alloc((size_t)2048 * 2048 * 2));
  float*  imgf   = (float*) (ws + alloc((size_t)2048 * 1024 * 4));
  bf16_t* tn_b   = (bf16_t*)(ws + alloc((size_t)16384 * 1024 * 2));
  bf16_t* in_b   = (bf16_t*)(ws + alloc((size_t)2048 * 1024 * 2));
  float*  temp   = (float*) (ws + alloc((size_t)16384 * 1024 * 4));
  bf16_t* Qb     = (bf16_t*)(ws + alloc((size_t)256 * 576 * 128 * 2));
  bf16_t* Kb     = (bf16_t*)(ws + alloc((size_t)256 * 576 * 128 * 2));
  bf16_t* Vtb    = (bf16_t*)(ws + alloc((size_t)256 * 576 * 128 * 2));
  bf16_t* featb  = (bf16_t*)(ws + alloc((size_t)18432 * 1024 * 2));
  float*  qf     = (float*) (ws + alloc((size_t)32 * 1024 * 4));
  float*  af     = (float*) (ws + alloc((size_t)32 * 1024 * 4));
  float*  akf    = (float*) (ws + alloc((size_t)18432 * 1024 * 4));
  float*  avf    = (float*) (ws + alloc((size_t)18432 * 1024 * 4));
  float*  w2     = (float*) (ws + alloc((size_t)18432 * 4));
  float*  attsb  = (float*) (ws + alloc((size_t)32 * 1024 * 4));
  (void)ws_size; (void)in_sizes; (void)n_in;

  auto blocks = [](int n) { return (n + 255) / 256; };

  // 1) weight transpose + bf16 convert
  k_transpose_cvt<<<blocks(1024*1024), 256, 0, stream>>>(W_que, Wq_t, 1024, 1024, 1024*1024);
  k_transpose_cvt<<<blocks(1024*1024), 256, 0, stream>>>(W_ans, Wa_t, 1024, 1024, 1024*1024);
  k_transpose_cvt<<<blocks(2048*2048), 256, 0, stream>>>(W_im1, Wi1_t, 2048, 2048, 2048*2048);
  k_transpose_cvt<<<blocks(2048*1024), 256, 0, stream>>>(W_im2, Wi2_t, 2048, 1024, 2048*1024);
  k_transpose_cvt<<<blocks(1024*1024), 256, 0, stream>>>(W_sq, Wsq_t, 1024, 128, 1024*1024);
  k_transpose_cvt<<<blocks(1024*1024), 256, 0, stream>>>(W_sk, Wsk_t, 1024, 128, 1024*1024);
  k_transpose_cvt<<<blocks(1024*1024), 256, 0, stream>>>(W_sv, Wsv_t, 1024, 128, 1024*1024);
  k_transpose_cvt<<<blocks(1024*1024), 256, 0, stream>>>(W_iq, Wiq_t, 1024, 128, 1024*1024);
  k_transpose_cvt<<<blocks(1024*1024), 256, 0, stream>>>(W_ik, Wik_t, 1024, 128, 1024*1024);
  k_transpose_cvt<<<blocks(1024*1024), 256, 0, stream>>>(W_iv, Wiv_t, 1024, 128, 1024*1024);
  k_transpose_cvt<<<blocks(1024*1024), 256, 0, stream>>>(W_kp, Wkp_t, 1024, 1024, 1024*1024);
  k_transpose_cvt<<<blocks(1024*1024), 256, 0, stream>>>(W_vp, Wvp_t, 1024, 1024, 1024*1024);

  // 2) activation converts
  k_cvt_bf16<<<blocks(32*1024), 256, 0, stream>>>(question, qb16, 32*1024);
  k_cvt_bf16<<<blocks(32*1024), 256, 0, stream>>>(answer,   ab16, 32*1024);
  k_cvt_bf16<<<blocks(2048*2048), 256, 0, stream>>>(images, imgs_b, 2048*2048);

  // 3) q / a projections   (M=32, N=1024)
  k_gemm_wmma<<<dim3(8, 1), 128, 0, stream>>>(qb16, Wq_t, b_que, qf, nullptr, 32, 1024, 1024, 0);
  k_gemm_wmma<<<dim3(8, 1), 128, 0, stream>>>(ab16, Wa_t, b_ans, af, nullptr, 32, 1024, 1024, 0);

  // 4) image MLP
  k_gemm_wmma<<<dim3(16, 32), 128, 0, stream>>>(imgs_b, Wi1_t, b_im1, nullptr, h1_b, 2048, 2048, 2048, 1);
  k_gemm_wmma<<<dim3(8, 32), 128, 0, stream>>>(h1_b,   Wi2_t, b_im2, imgf, nullptr, 2048, 1024, 2048, 0);

  // 5) LayerNorms -> bf16
  k_ln_bf16<<<16384, 256, 0, stream>>>(text, ln_g, ln_b, tn_b, 1024);
  k_ln_bf16<<<2048, 256, 0, stream>>>(imgf, ln_g, ln_b, in_b, 1024);

  // 6) Q/K/V projections + scatter into [B,H,L,KD] (and V transposed), masked-zeroed
  const int TOT_T = 32 * 512 * 1024, TOT_I = 32 * 64 * 1024;
  k_gemm_wmma<<<dim3(8, 256), 128, 0, stream>>>(tn_b, Wsq_t, b_sq, temp, nullptr, 16384, 1024, 1024, 0);
  k_scatter_qk<<<blocks(TOT_T), 256, 0, stream>>>(temp, Qb, text_len, 512, 0, TOT_T);
  k_gemm_wmma<<<dim3(8, 32), 128, 0, stream>>>(in_b, Wiq_t, b_iq, temp, nullptr, 2048, 1024, 1024, 0);
  k_scatter_qk<<<blocks(TOT_I), 256, 0, stream>>>(temp, Qb, img_len, 64, 512, TOT_I);

  k_gemm_wmma<<<dim3(8, 256), 128, 0, stream>>>(tn_b, Wsk_t, b_sk, temp, nullptr, 16384, 1024, 1024, 0);
  k_scatter_qk<<<blocks(TOT_T), 256, 0, stream>>>(temp, Kb, text_len, 512, 0, TOT_T);
  k_gemm_wmma<<<dim3(8, 32), 128, 0, stream>>>(in_b, Wik_t, b_ik, temp, nullptr, 2048, 1024, 1024, 0);
  k_scatter_qk<<<blocks(TOT_I), 256, 0, stream>>>(temp, Kb, img_len, 64, 512, TOT_I);

  k_gemm_wmma<<<dim3(8, 256), 128, 0, stream>>>(tn_b, Wsv_t, b_sv, temp, nullptr, 16384, 1024, 1024, 0);
  k_scatter_vT<<<blocks(TOT_T), 256, 0, stream>>>(temp, Vtb, text_len, 512, 0, TOT_T);
  k_gemm_wmma<<<dim3(8, 32), 128, 0, stream>>>(in_b, Wiv_t, b_iv, temp, nullptr, 2048, 1024, 1024, 0);
  k_scatter_vT<<<blocks(TOT_I), 256, 0, stream>>>(temp, Vtb, img_len, 64, 512, TOT_I);

  // 7) fused masked attention -> feat [B, L, H*VD] bf16
  k_attn_fused<<<dim3(256, 36), 32, 0, stream>>>(Qb, Kb, Vtb, featb);

  // 8) ak / av projections + sequence mask
  const int TOT_L = 32 * 576 * 1024;
  k_gemm_wmma<<<dim3(8, 288), 128, 0, stream>>>(featb, Wkp_t, b_kp, akf, nullptr, 18432, 1024, 1024, 0);
  k_mask_seq<<<blocks(TOT_L), 256, 0, stream>>>(akf, text_len, img_len, TOT_L);
  k_gemm_wmma<<<dim3(8, 288), 128, 0, stream>>>(featb, Wvp_t, b_vp, avf, nullptr, 18432, 1024, 1024, 0);
  k_mask_seq<<<blocks(TOT_L), 256, 0, stream>>>(avf, text_len, img_len, TOT_L);

  // 9) second-stage attention + head
  k_w2_dot<<<2304, 256, 0, stream>>>(qf, akf, w2);
  k_softmax_w2<<<32, 256, 0, stream>>>(w2);
  k_atts<<<32, 256, 0, stream>>>(w2, avf, attsb);
  k_final<<<1, 32, 0, stream>>>(qf, af, attsb, W_ap, b_ap, label, out, out_size);
}
